// Qwen3OmniMoeTransformerDecoderLayer_88424786690398
// MI455X (gfx1250) — compile-verified, matrix operand order checked by reading, not run on hardware
//
#include <hip/hip_runtime.h>
#include <hip/hip_bf16.h>

typedef __bf16 bf16;
typedef __attribute__((ext_vector_type(16))) __bf16 v16bf;
typedef __attribute__((ext_vector_type(8)))  float  v8f;
typedef unsigned int u32x4 __attribute__((ext_vector_type(4)));
typedef int i32x8 __attribute__((ext_vector_type(8)));
typedef int i32x4 __attribute__((ext_vector_type(4)));

#define S_  2048
#define H_  2048
#define HQ_ 16
#define HKV_ 4
#define D_  128
#define E_  16
#define I_  768

#if defined(__has_builtin)
#if __has_builtin(__builtin_amdgcn_tensor_load_to_lds) && \
    __has_builtin(__builtin_amdgcn_s_wait_tensorcnt)
#define HAVE_TDM 1
#endif
#endif
#ifndef HAVE_TDM
#define HAVE_TDM 0
#endif

// ---------------------------------------------------------------------------
// WMMA helper: D = A(16x32 bf16) * B(32x16 bf16) + C(16x16 f32)
// ---------------------------------------------------------------------------
__device__ __forceinline__ v8f wmma_bf(v16bf a, v16bf b, v8f c) {
  return __builtin_amdgcn_wmma_f32_16x16x32_bf16(false, a, false, b, (short)0, c,
                                                 false, false);
}

// Fragment load from a row-major matrix with explicit per-lane row (gather).
// ISA A-layout: lanes 0-15 hold row M with K = k0+{0..7,16..23}; lanes 16-31
// hold the same rows with K = k0+{8..15,24..31}. Both 8-element runs are
// contiguous -> two 16B loads. The B operand uses the same pattern on the
// *transposed* matrix (N x K), so every operand load is contiguous.
__device__ __forceinline__ v16bf frag_ld_row(const bf16* base, int ld, int row,
                                             int k0) {
  const int lane = threadIdx.x & 31;
  const bf16* p = base + (size_t)row * ld + k0 + ((lane >> 4) << 3);
  v16bf f;
#pragma unroll
  for (int i = 0; i < 8; ++i) f[i] = p[i];
#pragma unroll
  for (int i = 0; i < 8; ++i) f[i + 8] = p[16 + i];
  return f;
}

__device__ __forceinline__ v16bf frag_ld(const bf16* A, int ld, int row0, int k0) {
  const int lane = threadIdx.x & 31;
  return frag_ld_row(A, ld, row0 + (lane & 15), k0);
}

#if HAVE_TDM
// ---------------------------------------------------------------------------
// TDM: async 2-D tile load (tile_rows x tile_k bf16) from a (N x K) row-major
// bf16 matrix into LDS. Descriptor per CDNA5 ISA sec. 8 (D# groups 0/1).
// ---------------------------------------------------------------------------
__device__ __forceinline__ void tdm_load_2d(unsigned lds_off, const bf16* gaddr,
                                            int K, int N, int tile_k,
                                            int tile_rows) {
  unsigned long long ga = (unsigned long long)(uintptr_t)gaddr;
  u32x4 g0;
  g0[0] = 1u;                                               // count=1, user mode
  g0[1] = lds_off;                                          // lds_addr (bytes)
  g0[2] = (unsigned)ga;                                     // global_addr[31:0]
  g0[3] = ((unsigned)(ga >> 32) & 0x01FFFFFFu) | 0x80000000u;  // [56:32] | type=2
  i32x8 g1;
  g1[0] = 0x00010000;  // wg_mask=0, data_size=1 (2 bytes), no barrier/iter/pad
  g1[1] = (int)(((unsigned)K & 0xFFFFu) << 16);             // tensor_dim0 lo16
  g1[2] = (int)(((unsigned)K >> 16) | (((unsigned)N & 0xFFFFu) << 16));
  g1[3] = (int)(((unsigned)N >> 16) | ((unsigned)tile_k << 16));  // tile_dim0
  g1[4] = tile_rows;                                        // tile_dim1 (dim2=0)
  g1[5] = K;                                                // dim0_stride[31:0]
  g1[6] = 0;
  g1[7] = 0;
  i32x4 gz = {0, 0, 0, 0};
#if __clang_major__ >= 23
  i32x8 gz8 = {};
  __builtin_amdgcn_tensor_load_to_lds(g0, g1, gz, gz, gz8, 0);
#else
  __builtin_amdgcn_tensor_load_to_lds(g0, g1, gz, gz, 0);
#endif
}
#endif

// ---------------------------------------------------------------------------
// f32 -> bf16 conversion (plain)
// ---------------------------------------------------------------------------
__global__ void cvt_kernel(const float* __restrict__ src, bf16* __restrict__ dst,
                           size_t n) {
  size_t i = (size_t)blockIdx.x * blockDim.x + threadIdx.x;
  if (i < n) dst[i] = (bf16)src[i];
}

// ---------------------------------------------------------------------------
// f32 -> bf16 conversion with transpose through an LDS tile.
// src: matrix z of shape (K, N) f32; dst: (N, K) bf16.
// grid = (ceil(N/32), ceil(K/32), nmat), block = 256.
// ---------------------------------------------------------------------------
__global__ void __launch_bounds__(256) cvt_t_kernel(const float* __restrict__ src,
                                                    bf16* __restrict__ dst, int K,
                                                    int N) {
  __shared__ bf16 t[32][33];
  const size_t mo = (size_t)blockIdx.z * K * N;
  const int kb = blockIdx.y * 32, nb = blockIdx.x * 32;
  const int tx = threadIdx.x & 31, ty = threadIdx.x >> 5;
#pragma unroll
  for (int r = ty; r < 32; r += 8)
    if (kb + r < K && nb + tx < N)
      t[r][tx] = (bf16)src[mo + (size_t)(kb + r) * N + nb + tx];
  __syncthreads();
#pragma unroll
  for (int r = ty; r < 32; r += 8)
    if (nb + r < N && kb + tx < K)
      dst[mo + (size_t)(nb + r) * K + kb + tx] = t[tx][r];
}

// ---------------------------------------------------------------------------
// bf16 GEMM: C[M,N] = A[M,K] @ Bt[N,K]^T (+resid). Wave computes 32x32 via
// 4 accumulators. The 128x32 B tile is staged into LDS by the Tensor Data
// Mover (double-buffered, s_wait_tensorcnt pipelined); A streams from global.
// grid = (N/128, M/64), block = 256 (8 waves as 2x4 row x col tiles).
// ---------------------------------------------------------------------------
template <bool FUSE>
__global__ void __launch_bounds__(256) gemm_bf16_kernel(
    const bf16* __restrict__ A, const bf16* __restrict__ Bt,
    float* __restrict__ C, bf16* __restrict__ Cb, const float* __restrict__ resid,
    int M, int N, int K) {
#if HAVE_TDM
  __shared__ alignas(16) bf16 bstage[2][128 * 32];
#endif
  const int wave = threadIdx.x >> 5;
  const int lane = threadIdx.x & 31;
  const int row0 = blockIdx.y * 64 + (wave >> 2) * 32;
  const int colb = blockIdx.x * 128;
  const int colw = (wave & 3) * 32;  // column offset inside the 128-wide tile
#if HAVE_TDM
  const int wv = __builtin_amdgcn_readfirstlane(wave);  // scalar branch guard
  if (wv == 0)
    tdm_load_2d((unsigned)(uintptr_t)&bstage[0][0], Bt + (size_t)colb * K, K, N,
                32, 128);
#endif
  v8f a00 = {}, a01 = {}, a10 = {}, a11 = {};
  for (int k0 = 0; k0 < K; k0 += 32) {
#if HAVE_TDM
    const int cur = (k0 >> 5) & 1;
    if (wv == 0) {
      if (k0 + 32 < K) {
        tdm_load_2d((unsigned)(uintptr_t)&bstage[cur ^ 1][0],
                    Bt + (size_t)colb * K + (k0 + 32), K, N, 32, 128);
        __builtin_amdgcn_s_wait_tensorcnt(1);
      } else {
        __builtin_amdgcn_s_wait_tensorcnt(0);
      }
    }
    __syncthreads();
    const bf16* bsrc = &bstage[cur][0];
    v16bf bf0 = frag_ld(bsrc, 32, colw, 0);
    v16bf bf1 = frag_ld(bsrc, 32, colw + 16, 0);
#else
    v16bf bf0 = frag_ld(Bt, K, colb + colw, k0);
    v16bf bf1 = frag_ld(Bt, K, colb + colw + 16, k0);
#endif
    __builtin_prefetch(A + (size_t)row0 * K + k0 + 256, 0, 1);
    v16bf af0 = frag_ld(A, K, row0, k0);
    v16bf af1 = frag_ld(A, K, row0 + 16, k0);
    a00 = wmma_bf(af0, bf0, a00);
    a01 = wmma_bf(af0, bf1, a01);
    a10 = wmma_bf(af1, bf0, a10);
    a11 = wmma_bf(af1, bf1, a11);
#if HAVE_TDM
    __syncthreads();
#endif
  }
  const int rb = (lane >> 4) << 3;
  const int nn = lane & 15;
  v8f accs[4] = {a00, a01, a10, a11};
#pragma unroll
  for (int t = 0; t < 4; ++t) {
    const int r0 = row0 + (t >> 1) * 16 + rb;
    const int c0 = colb + colw + (t & 1) * 16 + nn;
#pragma unroll
    for (int i = 0; i < 8; ++i) {
      size_t idx = (size_t)(r0 + i) * N + c0;
      float v = accs[t][i];
      if (FUSE) {
        v += resid[idx];
        C[idx] = v;
        Cb[idx] = (bf16)v;
      } else {
        C[idx] = v;
      }
    }
  }
}

// ---------------------------------------------------------------------------
// Per-head RMSNorm + RoPE, f32 in -> bf16 out. One wave per (token, head) row.
// ---------------------------------------------------------------------------
__global__ void __launch_bounds__(256) normrope_kernel(
    const float* __restrict__ src, bf16* __restrict__ dst,
    const float* __restrict__ scale, int nh) {
  const int wave = threadIdx.x >> 5;
  const int lane = threadIdx.x & 31;
  const int row = blockIdx.x * 8 + wave;
  const int tok = row / nh;
  const float* x = src + (size_t)row * D_;
  float v[4];
  float ss = 0.f;
#pragma unroll
  for (int j = 0; j < 4; ++j) { v[j] = x[lane * 4 + j]; ss += v[j] * v[j]; }
#pragma unroll
  for (int off = 1; off < 32; off <<= 1) ss += __shfl_xor(ss, off, 32);
  const float inv = rsqrtf(ss * (1.0f / (float)D_) + 1e-6f);
  const float LOG_TH = 9.210340371976184f;  // ln(10000)
  bf16 out[4];
#pragma unroll
  for (int j = 0; j < 4; ++j) {
    const int e = lane * 4 + j;
    float xn = v[j] * inv * scale[e];
    float partner = __shfl_xor(xn, 16, 32);  // element e ^ 64
    const int jj = e & 63;
    float ang = (float)tok * __expf(-LOG_TH * (float)jj * (1.0f / 64.0f));
    float cs = __cosf(ang), sn = __sinf(ang);
    float r = (e < 64) ? (xn * cs - partner * sn) : (xn * cs + partner * sn);
    out[j] = (bf16)r;
  }
  bf16* d = dst + (size_t)row * D_ + lane * 4;
#pragma unroll
  for (int j = 0; j < 4; ++j) d[j] = out[j];
}

// ---------------------------------------------------------------------------
// Flash attention: one wave per (head, 16-row q tile). Online softmax in the
// WMMA C layout; P goes through LDS to become an A fragment for P@V.
// V comes in transposed as vt[HKV*D, S] so its fragments are contiguous.
// ---------------------------------------------------------------------------
__global__ void __launch_bounds__(32) attn_kernel(
    const bf16* __restrict__ qb, const bf16* __restrict__ kb,
    const bf16* __restrict__ vt, bf16* __restrict__ ob) {
  __shared__ bf16 pls[16 * 32];
  const int head = blockIdx.x;
  const int q0 = blockIdx.y * 16;
  const int lane = threadIdx.x & 31;
  const int kvh = head >> 2;  // HQ/HKV = 4
  const bf16* qbase = qb + (size_t)head * D_;      // ld = HQ*D
  const bf16* kbase = kb + (size_t)kvh * D_;       // ld = HKV*D
  const bf16* vbase = vt + (size_t)kvh * D_ * S_;  // (D, S) for this head

  v16bf qa[4];
#pragma unroll
  for (int c = 0; c < 4; ++c) qa[c] = frag_ld(qbase, HQ_ * D_, q0, c * 32);

  v8f o[8];
  float m[8], l[8];
#pragma unroll
  for (int c = 0; c < 8; ++c) o[c] = (v8f){};
#pragma unroll
  for (int i = 0; i < 8; ++i) { m[i] = -1e30f; l[i] = 0.f; }

  const float sc = 0.08838834764831845f;  // 1/sqrt(128)
  const int rb = (lane >> 4) << 3;
  const int kcol = lane & 15;

  for (int k0 = 0; k0 < q0 + 16; k0 += 32) {
    v8f s0 = {}, s1 = {};
#pragma unroll
    for (int c = 0; c < 4; ++c) {
      v16bf kf0 = frag_ld(kbase, HKV_ * D_, k0, c * 32);
      v16bf kf1 = frag_ld(kbase, HKV_ * D_, k0 + 16, c * 32);
      s0 = wmma_bf(qa[c], kf0, s0);
      s1 = wmma_bf(qa[c], kf1, s1);
    }
    float fs[8];
#pragma unroll
    for (int i = 0; i < 8; ++i) {
      const int qr = q0 + rb + i;
      float a0 = (k0 + kcol <= qr) ? s0[i] * sc : -1e30f;
      float a1 = (k0 + 16 + kcol <= qr) ? s1[i] * sc : -1e30f;
      float mx = fmaxf(a0, a1);
#pragma unroll
      for (int off = 1; off < 16; off <<= 1) mx = fmaxf(mx, __shfl_xor(mx, off, 32));
      float nm = fmaxf(m[i], mx);
      float f = __expf(m[i] - nm);
      m[i] = nm;
      float p0 = __expf(a0 - nm), p1 = __expf(a1 - nm);
      float rs = p0 + p1;
#pragma unroll
      for (int off = 1; off < 16; off <<= 1) rs += __shfl_xor(rs, off, 32);
      l[i] = l[i] * f + rs;
      fs[i] = f;
      pls[(rb + i) * 32 + kcol] = (bf16)p0;
      pls[(rb + i) * 32 + 16 + kcol] = (bf16)p1;
    }
    __syncthreads();
#pragma unroll
    for (int c = 0; c < 8; ++c)
#pragma unroll
      for (int i = 0; i < 8; ++i) o[c][i] *= fs[i];
    v16bf pf = frag_ld(pls, 32, 0, 0);
#pragma unroll
    for (int c = 0; c < 8; ++c) {
      v16bf vf = frag_ld(vbase, S_, c * 16, k0);  // rows = d, k = keys
      o[c] = wmma_bf(pf, vf, o[c]);
    }
    __syncthreads();
  }
#pragma unroll
  for (int c = 0; c < 8; ++c) {
#pragma unroll
    for (int i = 0; i < 8; ++i) {
      float v = o[c][i] / l[i];
      size_t row = (size_t)(q0 + rb + i);
      ob[row * (HQ_ * D_) + head * D_ + c * 16 + kcol] = (bf16)v;
    }
  }
}

// ---------------------------------------------------------------------------
// Router GEMM: logits[S,16] = hb[S,H] @ rwt[16,H]^T. One wave per 16 tokens.
// ---------------------------------------------------------------------------
__global__ void __launch_bounds__(32) router_kernel(
    const bf16* __restrict__ hb, const bf16* __restrict__ rwt,
    float* __restrict__ logits) {
  const int lane = threadIdx.x & 31;
  const int row0 = blockIdx.x * 16;
  v8f acc = {};
  for (int k0 = 0; k0 < H_; k0 += 32) {
    v16bf a = frag_ld(hb, H_, row0, k0);
    v16bf b = frag_ld(rwt, H_, 0, k0);
    acc = wmma_bf(a, b, acc);
  }
  const int n = lane & 15;
  const int rbt = row0 + ((lane >> 4) << 3);
#pragma unroll
  for (int i = 0; i < 8; ++i) logits[(size_t)(rbt + i) * E_ + n] = acc[i];
}

__global__ void zero_cnt_kernel(int* cnt) {
  if (threadIdx.x < E_) cnt[threadIdx.x] = 0;
}

__global__ void topk_kernel(const float* __restrict__ logits, int* __restrict__ cnt,
                            int* __restrict__ tok) {
  int t = blockIdx.x * blockDim.x + threadIdx.x;
  if (t >= S_) return;
  const float* lg = logits + (size_t)t * E_;
  int i1 = 0;
  float b1 = lg[0];
#pragma unroll
  for (int e = 1; e < E_; ++e)
    if (lg[e] > b1) { b1 = lg[e]; i1 = e; }
  int i2 = -1;
  float b2 = -1e30f;
#pragma unroll
  for (int e = 0; e < E_; ++e)
    if (e != i1 && lg[e] > b2) { b2 = lg[e]; i2 = e; }
  int s1 = atomicAdd(&cnt[i1], 1);
  tok[i1 * S_ + s1] = t;
  int s2 = atomicAdd(&cnt[i2], 1);
  tok[i2 * S_ + s2] = t;
}

// ---------------------------------------------------------------------------
// MoE up projection (gathered rows) with fused bias + SiLU -> bf16.
// wupt: per expert (I, H). grid = (I/256, E), block 256; wave = 16 x 32 tile.
// ---------------------------------------------------------------------------
__global__ void __launch_bounds__(256) moe_up_kernel(
    const bf16* __restrict__ hb, const bf16* __restrict__ wupt,
    const float* __restrict__ bup, const int* __restrict__ cnt,
    const int* __restrict__ tok, bf16* __restrict__ upbuf) {
  const int e = blockIdx.y;
  const int wave = threadIdx.x >> 5;
  const int lane = threadIdx.x & 31;
  const int n0 = blockIdx.x * 256 + wave * 32;
  const int c = cnt[e];
  const bf16* W = wupt + (size_t)e * I_ * H_;
  const int* tl = tok + e * S_;
  for (int m0 = 0; m0 < c; m0 += 16) {
    int slot = m0 + (lane & 15);
    int trow = tl[slot < c ? slot : c - 1];
    v8f acc0 = {}, acc1 = {};
    for (int k0 = 0; k0 < H_; k0 += 32) {
      __builtin_prefetch(W + (size_t)n0 * H_ + k0 + 256, 0, 1);
      v16bf a = frag_ld_row(hb, H_, trow, k0);
      v16bf b0 = frag_ld(W, H_, n0, k0);
      v16bf b1 = frag_ld(W, H_, n0 + 16, k0);
      acc0 = wmma_bf(a, b0, acc0);
      acc1 = wmma_bf(a, b1, acc1);
    }
    const int rbt = m0 + ((lane >> 4) << 3);
    v8f accs[2] = {acc0, acc1};
#pragma unroll
    for (int t = 0; t < 2; ++t) {
      const int n = n0 + t * 16 + (lane & 15);
      const float bias = bup[e * I_ + n];
#pragma unroll
      for (int i = 0; i < 8; ++i) {
        int sr = rbt + i;
        if (sr < c) {
          float x = accs[t][i] + bias;
          float sl = x / (1.0f + __expf(-x));
          upbuf[((size_t)e * S_ + sr) * I_ + n] = (bf16)sl;
        }
      }
    }
  }
}

// ---------------------------------------------------------------------------
// MoE down projection + bias, scattered atomicAdd into out (binary routing).
// wdnt: per expert (H, I). grid = (H/256, E), block 256.
// ---------------------------------------------------------------------------
__global__ void __launch_bounds__(256) moe_down_kernel(
    const bf16* __restrict__ upbuf, const bf16* __restrict__ wdnt,
    const float* __restrict__ bdn, const int* __restrict__ cnt,
    const int* __restrict__ tok, float* __restrict__ out) {
  const int e = blockIdx.y;
  const int wave = threadIdx.x >> 5;
  const int lane = threadIdx.x & 31;
  const int n0 = blockIdx.x * 256 + wave * 32;
  const int c = cnt[e];
  const bf16* A = upbuf + (size_t)e * S_ * I_;
  const bf16* W = wdnt + (size_t)e * H_ * I_;
  const int* tl = tok + e * S_;
  for (int m0 = 0; m0 < c; m0 += 16) {
    v8f acc0 = {}, acc1 = {};
    for (int k0 = 0; k0 < I_; k0 += 32) {
      v16bf a = frag_ld(A, I_, m0, k0);
      v16bf b0 = frag_ld(W, I_, n0, k0);
      v16bf b1 = frag_ld(W, I_, n0 + 16, k0);
      acc0 = wmma_bf(a, b0, acc0);
      acc1 = wmma_bf(a, b1, acc1);
    }
    const int rbt = m0 + ((lane >> 4) << 3);
    v8f accs[2] = {acc0, acc1};
#pragma unroll
    for (int t = 0; t < 2; ++t) {
      const int n = n0 + t * 16 + (lane & 15);
      const float bias = bdn[e * H_ + n];
#pragma unroll
      for (int i = 0; i < 8; ++i) {
        int sr = rbt + i;
        if (sr < c) {
          int tk = tl[sr];
          atomicAdd(&out[(size_t)tk * H_ + n], accs[t][i] + bias);
        }
      }
    }
  }
}

// ---------------------------------------------------------------------------
// Host launcher
// ---------------------------------------------------------------------------
extern "C" void kernel_launch(void* const* d_in, const int* in_sizes, int n_in,
                              void* d_out, int out_size, void* d_ws, size_t ws_size,
                              hipStream_t stream) {
  const float* hidden  = (const float*)d_in[0];
  const float* wq      = (const float*)d_in[1];
  const float* wk      = (const float*)d_in[2];
  const float* wv      = (const float*)d_in[3];
  const float* wo      = (const float*)d_in[4];
  const float* q_scale = (const float*)d_in[5];
  const float* k_scale = (const float*)d_in[6];
  const float* rw      = (const float*)d_in[7];
  const float* w_up    = (const float*)d_in[8];
  const float* b_up    = (const float*)d_in[9];
  const float* w_down  = (const float*)d_in[10];
  const float* b_down  = (const float*)d_in[11];

  float* out_h      = (float*)d_out;                    // (S, H)
  float* out_logits = (float*)d_out + (size_t)S_ * H_;  // (S, E)

  char* ws = (char*)d_ws;
  size_t off = 0;
  auto carve = [&](size_t bytes) -> char* {
    off = (off + 255) & ~(size_t)255;
    char* p = ws + off;
    off += bytes;
    return p;
  };
  bf16* xb    = (bf16*)carve((size_t)S_ * H_ * 2);
  bf16* wq_t  = (bf16*)carve((size_t)HQ_ * D_ * H_ * 2);   // (HQ*D, H)
  bf16* wk_t  = (bf16*)carve((size_t)HKV_ * D_ * H_ * 2);  // (HKV*D, H)
  bf16* wv_t  = (bf16*)carve((size_t)HKV_ * D_ * H_ * 2);
  bf16* wo_t  = (bf16*)carve((size_t)H_ * HQ_ * D_ * 2);   // (H, HQ*D)
  bf16* rw_t  = (bf16*)carve((size_t)E_ * H_ * 2);         // (E, H)
  bf16* wup_t = (bf16*)carve((size_t)E_ * I_ * H_ * 2);    // (E, I, H)
  bf16* wdn_t = (bf16*)carve((size_t)E_ * H_ * I_ * 2);    // (E, H, I)
  float* qf   = (float*)carve((size_t)S_ * HQ_ * D_ * 4);
  float* kf   = (float*)carve((size_t)S_ * HKV_ * D_ * 4);
  float* vf   = (float*)carve((size_t)S_ * HKV_ * D_ * 4);
  bf16* qb    = (bf16*)carve((size_t)S_ * HQ_ * D_ * 2);
  bf16* kb2   = (bf16*)carve((size_t)S_ * HKV_ * D_ * 2);
  bf16* vt    = (bf16*)carve((size_t)HKV_ * D_ * S_ * 2);  // (HKV*D, S)
  bf16* attnb = (bf16*)carve((size_t)S_ * HQ_ * D_ * 2);
  bf16* hb    = (bf16*)carve((size_t)S_ * H_ * 2);
  int* cnt    = (int*)carve(E_ * 4);
  int* tok    = (int*)carve((size_t)E_ * S_ * 4);
  bf16* upbuf = (bf16*)carve((size_t)E_ * S_ * I_ * 2);
  (void)ws_size; (void)n_in; (void)in_sizes; (void)out_size;

  // 1) precision conversion (weights transposed so B-fragments are contiguous)
  cvt_kernel<<<dim3((S_ * H_ + 255) / 256), 256, 0, stream>>>(hidden, xb,
                                                              (size_t)S_ * H_);
  cvt_t_kernel<<<dim3(HQ_ * D_ / 32, H_ / 32, 1), 256, 0, stream>>>(wq, wq_t, H_,
                                                                    HQ_ * D_);
  cvt_t_kernel<<<dim3(HKV_ * D_ / 32, H_ / 32, 1), 256, 0, stream>>>(wk, wk_t, H_,
                                                                     HKV_ * D_);
  cvt_t_kernel<<<dim3(HKV_ * D_ / 32, H_ / 32, 1), 256, 0, stream>>>(wv, wv_t, H_,
                                                                     HKV_ * D_);
  cvt_t_kernel<<<dim3(H_ / 32, HQ_ * D_ / 32, 1), 256, 0, stream>>>(wo, wo_t,
                                                                    HQ_ * D_, H_);
  cvt_t_kernel<<<dim3(1, H_ / 32, 1), 256, 0, stream>>>(rw, rw_t, H_, E_);
  cvt_t_kernel<<<dim3(I_ / 32, H_ / 32, E_), 256, 0, stream>>>(w_up, wup_t, H_, I_);
  cvt_t_kernel<<<dim3(H_ / 32, I_ / 32, E_), 256, 0, stream>>>(w_down, wdn_t, I_,
                                                               H_);

  // 2) QKV projections
  gemm_bf16_kernel<false><<<dim3(HQ_ * D_ / 128, S_ / 64), 256, 0, stream>>>(
      xb, wq_t, qf, nullptr, nullptr, S_, HQ_ * D_, H_);
  gemm_bf16_kernel<false><<<dim3(HKV_ * D_ / 128, S_ / 64), 256, 0, stream>>>(
      xb, wk_t, kf, nullptr, nullptr, S_, HKV_ * D_, H_);
  gemm_bf16_kernel<false><<<dim3(HKV_ * D_ / 128, S_ / 64), 256, 0, stream>>>(
      xb, wv_t, vf, nullptr, nullptr, S_, HKV_ * D_, H_);

  // 3) RMSNorm + RoPE (q, k); V transposed to (HKV*D, S) bf16
  normrope_kernel<<<dim3(S_ * HQ_ / 8), 256, 0, stream>>>(qf, qb, q_scale, HQ_);
  normrope_kernel<<<dim3(S_ * HKV_ / 8), 256, 0, stream>>>(kf, kb2, k_scale, HKV_);
  cvt_t_kernel<<<dim3(HKV_ * D_ / 32, S_ / 32, 1), 256, 0, stream>>>(vf, vt, S_,
                                                                     HKV_ * D_);

  // 4) causal GQA flash attention
  attn_kernel<<<dim3(HQ_, S_ / 16), 32, 0, stream>>>(qb, kb2, vt, attnb);

  // 5) output projection + residual: h (f32 -> d_out) and hb (bf16)
  gemm_bf16_kernel<true><<<dim3(H_ / 128, S_ / 64), 256, 0, stream>>>(
      attnb, wo_t, out_h, hb, hidden, S_, H_, HQ_ * D_);

  // 6) router logits + top-2 routing
  router_kernel<<<dim3(S_ / 16), 32, 0, stream>>>(hb, rw_t, out_logits);
  zero_cnt_kernel<<<1, 32, 0, stream>>>(cnt);
  topk_kernel<<<dim3(S_ / 256), 256, 0, stream>>>(out_logits, cnt, tok);

  // 7) MoE grouped GEMMs; down-proj atomically adds into h in d_out
  moe_up_kernel<<<dim3(I_ / 256, E_), 256, 0, stream>>>(hb, wup_t, b_up, cnt, tok,
                                                        upbuf);
  moe_down_kernel<<<dim3(H_ / 256, E_), 256, 0, stream>>>(upbuf, wdn_t, b_down, cnt,
                                                          tok, out_h);
}
